// GraphSAGE_28896539967646
// MI455X (gfx1250) — matrix operand order repaired and checked by reading, not compiled, not run on hardware
//
#include <hip/hip_runtime.h>
#include <hip/hip_bf16.h>

typedef __attribute__((ext_vector_type(16))) __bf16 v16bf;
typedef __attribute__((ext_vector_type(8)))  __bf16 v8bf;
typedef __attribute__((ext_vector_type(8)))  float  v8f;

#define GN 100000
#define GE 1600000

// ---------------------------------------------------------------- utilities
__global__ __launch_bounds__(256) void k_zero_f32(float* p, long long n) {
    long long i = (long long)blockIdx.x * blockDim.x + threadIdx.x;
    if (i < n) p[i] = 0.0f;
}

// Build transposed bf16 weight block: Bt[n][k], k<128 -> Wl[k][n], else Wr[k-128][n]
__global__ __launch_bounds__(256) void k_build_bt(const float* __restrict__ Wl,
                                                  const float* __restrict__ Wr,
                                                  __bf16* __restrict__ Bt, int ncols) {
    int i = blockIdx.x * blockDim.x + threadIdx.x;
    if (i >= ncols * 256) return;
    int n = i >> 8;
    int k = i & 255;
    float v = (k < 128) ? Wl[k * ncols + n] : Wr[(k - 128) * ncols + n];
    Bt[n * 256 + k] = (__bf16)v;
}

// One wave per edge: lanes cover 128 features (4 per lane). Coalesced gather of
// src row (512B) + contiguous f32 atomics into dst row (L2-resident).
template <typename SrcT>
__global__ __launch_bounds__(256) void k_scatter(const long long* __restrict__ ei,
                                                 const SrcT* __restrict__ feat,
                                                 float* __restrict__ msg,
                                                 float* __restrict__ deg) {
    long long t = (long long)blockIdx.x * blockDim.x + threadIdx.x;
    long long e = t >> 5;
    int lane = (int)(t & 31);
    if (e >= GE) return;
    int src = (int)ei[e];
    int dst = (int)ei[(long long)GE + e];
    if (lane == 0) atomicAdd(&deg[dst], 1.0f);
    const SrcT* srow = feat + (long long)src * 128;
    float* drow = msg + (long long)dst * 128;
#pragma unroll
    for (int j = 0; j < 4; ++j) {
        int f = lane + j * 32;
        atomicAdd(&drow[f], (float)srow[f]);
    }
}

// A = [ msg/deg | self ] converted to bf16, row stride 256.
template <typename SelfT>
__global__ __launch_bounds__(256) void k_build_a(const float* __restrict__ msg,
                                                 const float* __restrict__ deg,
                                                 const SelfT* __restrict__ self,
                                                 __bf16* __restrict__ A) {
    long long i = (long long)blockIdx.x * blockDim.x + threadIdx.x;
    if (i >= (long long)GN * 128) return;
    long long node = i >> 7;
    int f = (int)(i & 127);
    float d = deg[node];
    d = d > 1.0f ? d : 1.0f;
    A[node * 256 + f]       = (__bf16)(msg[i] / d);
    A[node * 256 + 128 + f] = (__bf16)((float)self[node * 128 + f]);
}

__device__ __forceinline__ unsigned drop_hash(unsigned v) {
    v ^= 42u * 0x9E3779B9u;
    v *= 0x85EBCA6Bu; v ^= v >> 13;
    v *= 0xC2B2AE35u; v ^= v >> 16;
    return v;
}

__device__ __forceinline__ v16bf load_afrag(const __bf16* arow, int k0, int half) {
    v8bf lo = *(const v8bf*)(arow + k0 + half * 16);
    v8bf hi = *(const v8bf*)(arow + k0 + half * 16 + 8);
    return __builtin_shufflevector(lo, hi, 0, 1, 2, 3, 4, 5, 6, 7,
                                   8, 9, 10, 11, 12, 13, 14, 15);
}

// ---------------------------------------------------------------- WMMA GEMM
// C[M x NCOLS] = A[M x 256] @ B[256 x NCOLS] (+bias, optional ReLU+dropout).
// B supplied pre-transposed bf16: Bt[NCOLS][256]; staged whole into LDS.
// Block = 256 threads = 8 waves; wave w owns a 32-row slab (two 16-row M-tiles)
// so each B fragment read from LDS feeds TWO WMMAs (halves ds traffic).
template <int NCOLS, bool L1>
__global__ __launch_bounds__(256) void k_sage_gemm(const __bf16* __restrict__ A,
                                                   const __bf16* __restrict__ Bt,
                                                   const float* __restrict__ bias,
                                                   void* __restrict__ outp, int M) {
    __shared__ alignas(32) __bf16 Bs[NCOLS * 256];

    // cooperative 16B-chunk copy of the full B tile into LDS
    {
        const uint4* g = (const uint4*)Bt;
        uint4* l = (uint4*)Bs;
        const int chunks = NCOLS * 512 / 16;
        for (int c = threadIdx.x; c < chunks; c += 256) l[c] = g[c];
    }
    __syncthreads();

    const int wave = threadIdx.x >> 5;
    const int lane = threadIdx.x & 31;
    const int half = lane >> 4;   // K-half select for A/B fragments
    const int l15  = lane & 15;   // M (A) / N (B,C) within tile
    const long long rowBase = (long long)blockIdx.x * 256 + wave * 32;
    if (rowBase >= M) return;

    constexpr int NT = NCOLS / 16;
    v8f acc[2][NT];
    const v8f vzero = {0.f, 0.f, 0.f, 0.f, 0.f, 0.f, 0.f, 0.f};
#pragma unroll
    for (int r = 0; r < 2; ++r)
#pragma unroll
        for (int t = 0; t < NT; ++t) acc[r][t] = vzero;

    const __bf16* arow0 = A + (rowBase + l15) * 256;
    const __bf16* arow1 = arow0 + 16 * 256;

#pragma unroll
    for (int k0 = 0; k0 < 256; k0 += 32) {
        // Two A fragments (rows rowBase.. and rowBase+16..), 16x32 layout:
        // lane<16 -> K k0..k0+15 contiguous; lane>=16 -> k0+16..k0+31
        v16bf a0 = load_afrag(arow0, k0, half);
        v16bf a1 = load_afrag(arow1, k0, half);
#pragma unroll
        for (int t = 0; t < NT; ++t) {
            // B fragment (32x16): lane<16 -> col l15, K k0..k0+15 contiguous
            const v16bf bfrag =
                *(const v16bf*)(Bs + (t * 16 + l15) * 256 + k0 + half * 16);
            acc[0][t] = __builtin_amdgcn_wmma_f32_16x16x32_bf16(
                false, a0, false, bfrag, (short)0, acc[0][t], false, false);
            acc[1][t] = __builtin_amdgcn_wmma_f32_16x16x32_bf16(
                false, a1, false, bfrag, (short)0, acc[1][t], false, false);
        }
    }

    // Epilogue: C VGPR j -> row j + 8*half (+16*r slab), col l15 (+16*t)
#pragma unroll
    for (int r = 0; r < 2; ++r) {
#pragma unroll
        for (int t = 0; t < NT; ++t) {
            int col = t * 16 + l15;
            float bv = bias[col];
#pragma unroll
            for (int j = 0; j < 8; ++j) {
                long long row = rowBase + r * 16 + half * 8 + j;
                float v = acc[r][t][j] + bv;
                if constexpr (L1) {
                    v = v > 0.f ? v : 0.f;  // ReLU
                    unsigned h = drop_hash((unsigned)(row * NCOLS + col));
                    v = (h < 1288490189u) ? v * (1.0f / 0.3f) : 0.0f;  // p_keep=0.3
                    ((__bf16*)outp)[row * NCOLS + col] = (__bf16)v;
                } else {
                    ((float*)outp)[row * NCOLS + col] = v;
                }
            }
        }
    }
}

// ---------------------------------------------------------------- launcher
extern "C" void kernel_launch(void* const* d_in, const int* in_sizes, int n_in,
                              void* d_out, int out_size, void* d_ws, size_t ws_size,
                              hipStream_t stream) {
    const float*     x   = (const float*)d_in[0];
    const long long* ei  = (const long long*)d_in[1];
    const float*     W1l = (const float*)d_in[2];
    const float*     W1r = (const float*)d_in[3];
    const float*     b1  = (const float*)d_in[4];
    const float*     W2l = (const float*)d_in[5];
    const float*     W2r = (const float*)d_in[6];
    const float*     b2  = (const float*)d_in[7];
    float* out = (float*)d_out;

    // workspace layout (256B-aligned slices)
    char* ws = (char*)d_ws;
    size_t off = 0;
    auto take = [&](size_t bytes) {
        void* p = ws + off;
        off += (bytes + 255) & ~size_t(255);
        return p;
    };
    float*   msg = (float*)take((size_t)GN * 128 * 4 + (size_t)GN * 4);  // msg|deg contiguous
    float*   deg = msg + (size_t)GN * 128;
    __bf16*  Abuf = (__bf16*)take((size_t)GN * 256 * 2);
    __bf16*  hbuf = (__bf16*)take((size_t)GN * 128 * 2);
    __bf16*  Bt1  = (__bf16*)take((size_t)128 * 256 * 2);
    __bf16*  Bt2  = (__bf16*)take((size_t)64 * 256 * 2);

    const long long nMsg   = (long long)GN * 129;           // msg + deg
    const long long nFeat  = (long long)GN * 128;
    const long long nEdgeT = (long long)GE * 32;
    const int gemmBlocks   = (GN + 255) / 256;

    // weight conversion (once per call; independent of aggregation)
    k_build_bt<<<(128 * 256 + 255) / 256, 256, 0, stream>>>(W1l, W1r, Bt1, 128);
    k_build_bt<<<(64 * 256 + 255) / 256, 256, 0, stream>>>(W2l, W2r, Bt2, 64);

    // ---- layer 1 ----
    k_zero_f32<<<(int)((nMsg + 255) / 256), 256, 0, stream>>>(msg, nMsg);
    k_scatter<float><<<(int)((nEdgeT + 255) / 256), 256, 0, stream>>>(ei, x, msg, deg);
    k_build_a<float><<<(int)((nFeat + 255) / 256), 256, 0, stream>>>(msg, deg, x, Abuf);
    k_sage_gemm<128, true><<<gemmBlocks, 256, 0, stream>>>(Abuf, Bt1, b1, hbuf, GN);

    // ---- layer 2 ----
    k_zero_f32<<<(int)((nMsg + 255) / 256), 256, 0, stream>>>(msg, nMsg);
    k_scatter<__bf16><<<(int)((nEdgeT + 255) / 256), 256, 0, stream>>>(ei, hbuf, msg, deg);
    k_build_a<__bf16><<<(int)((nFeat + 255) / 256), 256, 0, stream>>>(msg, deg, hbuf, Abuf);
    k_sage_gemm<64, false><<<gemmBlocks, 256, 0, stream>>>(Abuf, Bt2, b2, out, GN);
}
